// AttentionFlowLayer_22419729285215
// MI455X (gfx1250) — matrix operand order repaired and checked by reading, not compiled, not run on hardware
//
#include <hip/hip_runtime.h>
#include <hip/hip_bf16.h>

// AttentionFlowLayer: out[b] = (H[b]*w3) @ U[b]^T + (H[b]@w1)[:,None] + (U[b]@w2)[None,:]
// B=8, T=2048, D=128, fp32 in/out.
// Batched GEMM via split-FP16 WMMA: x = hi + lo (f16 each, ~22 mantissa bits combined),
// A@B ~= Ahi*Bhi + Ahi*Blo + Alo*Bhi using V_WMMA_F32_16X16X32_F16 (3 passes, K=32/inst).
// Relative error ~1e-6 vs fp32; ~2.7x the math rate of the V_WMMA_F32_16X16X4_F32 path,
// targeting the 134MB-output-write HBM floor (~5.8us @ 23.3 TB/s).

typedef __attribute__((ext_vector_type(4)))  float    v4f;
typedef __attribute__((ext_vector_type(8)))  float    v8f;
typedef __attribute__((ext_vector_type(4)))  _Float16 v4h;
typedef __attribute__((ext_vector_type(8)))  _Float16 v8h;
typedef __attribute__((ext_vector_type(16))) _Float16 v16h;

#define TDIM 2048
#define DDIM 128
#define TILE 128
#define HSTR 136   // half-precision row stride: 272B == 4 banks mod 64 -> conflict-free frags

__global__ __launch_bounds__(256) void flow_gemm_f16split_kernel(
    const float* __restrict__ H, const float* __restrict__ U,
    const float* __restrict__ W, float* __restrict__ out)
{
    extern __shared__ char smem[];
    float*    shL = (float*)smem;            // 128 row biases (s_h)
    float*    suL = shL + TILE;              // 128 col biases (s_u)
    _Float16* Hhi = (_Float16*)(suL + TILE); // 128 x HSTR
    _Float16* Hlo = Hhi + TILE * HSTR;
    _Float16* Uhi = Hlo + TILE * HSTR;
    _Float16* Ulo = Uhi + TILE * HSTR;

    const int t  = threadIdx.x;
    const int b  = blockIdx.z;
    const int m0 = blockIdx.y * TILE;
    const int n0 = blockIdx.x * TILE;

    const float* __restrict__ Hb = H + ((size_t)b * TDIM + m0) * DDIM;
    const float* __restrict__ Ub = U + ((size_t)b * TDIM + n0) * DDIM;

    // ---- Phase 1: one row per thread (wave-uniform split: waves 0-3 -> H, 4-7 -> U).
    // Stream the row, accumulate bias dot deterministically, fold w3 into H before the
    // f16 hi/lo split, store packed f16 tiles to LDS.
    {
        const int  r   = t & 127;
        const bool isH = (t < 128);
        const float* __restrict__ src = isH ? (Hb + r * DDIM) : (Ub + r * DDIM);
        const float* __restrict__ wA  = isH ? W : (W + DDIM);      // w1 or w2 (scalar loads)
        _Float16* dHi = isH ? Hhi : Uhi;
        _Float16* dLo = isH ? Hlo : Ulo;

        float bias = 0.f;
        #pragma unroll 4
        for (int i = 0; i < 32; ++i) {
            v4f x  = *(const v4f*)(src + i * 4);
            v4f wa = *(const v4f*)(wA + i * 4);
            bias += x.x * wa.x + x.y * wa.y + x.z * wa.z + x.w * wa.w;
            v4f y = x;
            if (isH) {
                v4f w3c = *(const v4f*)(W + 2 * DDIM + i * 4);
                y *= w3c;
            }
            v4h hi4, lo4;
            #pragma unroll
            for (int j = 0; j < 4; ++j) {
                float    xf = y[j];
                _Float16 h  = (_Float16)xf;          // RNE, 11 mantissa bits
                hi4[j] = h;
                lo4[j] = (_Float16)(xf - (float)h);  // next ~11 bits
            }
            *(v4h*)(dHi + r * HSTR + i * 4) = hi4;
            *(v4h*)(dLo + r * HSTR + i * 4) = lo4;
        }
        if (isH) shL[r] = bias; else suL[r] = bias;
    }
    __syncthreads();

    // ---- Phase 2: WMMA f16-split GEMM. Wave grid 4(M) x 2(N); each wave: 32x64 = 2x4 tiles.
    const int lane = t & 31;
    const int wave = t >> 5;
    const int wm   = wave & 3;       // 32-row block
    const int wn   = wave >> 2;      // 64-col block
    const int lr   = lane & 15;
    const int ksel = lane >> 4;      // half-wave K selector per ISA 16-bit layouts

    v8f acc[2][4];
    #pragma unroll
    for (int mt = 0; mt < 2; ++mt)
        #pragma unroll
        for (int nt = 0; nt < 4; ++nt)
            acc[mt][nt] = {};

    const int aRow = wm * 32 + lr;   // +16 for mt=1
    const int bRow = wn * 64 + lr;   // +16*nt

    #pragma unroll
    for (int kc = 0; kc < DDIM; kc += 32) {
        // A fragments: lane<16 holds K{kc..kc+7, kc+16..kc+23}; lane>=16 offset +8.
        const int kA0 = kc + ksel * 8;
        v16h Ahi[2], Alo[2];
        #pragma unroll
        for (int mt = 0; mt < 2; ++mt) {
            const _Float16* ph = Hhi + (aRow + mt * 16) * HSTR;
            const _Float16* pl = Hlo + (aRow + mt * 16) * HSTR;
            v8h h0 = *(const v8h*)(ph + kA0);
            v8h h1 = *(const v8h*)(ph + kA0 + 16);
            v8h l0 = *(const v8h*)(pl + kA0);
            v8h l1 = *(const v8h*)(pl + kA0 + 16);
            Ahi[mt] = __builtin_shufflevector(h0, h1, 0,1,2,3,4,5,6,7,8,9,10,11,12,13,14,15);
            Alo[mt] = __builtin_shufflevector(l0, l1, 0,1,2,3,4,5,6,7,8,9,10,11,12,13,14,15);
        }
        // B fragments: lane<16 holds K{kc..kc+7, kc+16..kc+23}; lane>=16 offset +8.
        const int kB0 = kc + ksel * 8;
        #pragma unroll
        for (int nt = 0; nt < 4; ++nt) {
            const _Float16* ph = Uhi + (bRow + nt * 16) * HSTR;
            const _Float16* pl = Ulo + (bRow + nt * 16) * HSTR;
            v8h h0 = *(const v8h*)(ph + kB0);
            v8h h1 = *(const v8h*)(ph + kB0 + 16);
            v8h l0 = *(const v8h*)(pl + kB0);
            v8h l1 = *(const v8h*)(pl + kB0 + 16);
            v16h Bhi = __builtin_shufflevector(h0, h1, 0,1,2,3,4,5,6,7,8,9,10,11,12,13,14,15);
            v16h Blo = __builtin_shufflevector(l0, l1, 0,1,2,3,4,5,6,7,8,9,10,11,12,13,14,15);
            #pragma unroll
            for (int mt = 0; mt < 2; ++mt) {
                acc[mt][nt] = __builtin_amdgcn_wmma_f32_16x16x32_f16(false, Ahi[mt], false, Bhi,
                                                                     (short)0, acc[mt][nt], false, false);
                acc[mt][nt] = __builtin_amdgcn_wmma_f32_16x16x32_f16(false, Ahi[mt], false, Blo,
                                                                     (short)0, acc[mt][nt], false, false);
                acc[mt][nt] = __builtin_amdgcn_wmma_f32_16x16x32_f16(false, Alo[mt], false, Bhi,
                                                                     (short)0, acc[mt][nt], false, false);
            }
        }
    }

    // ---- Epilogue: add row/col biases, coalesced fp32 stores ----
    float su_v[4];
    #pragma unroll
    for (int nt = 0; nt < 4; ++nt) su_v[nt] = suL[wn * 64 + nt * 16 + lr];

    float* __restrict__ outb = out + (size_t)b * TDIM * TDIM;
    #pragma unroll
    for (int mt = 0; mt < 2; ++mt) {
        const int mloc = wm * 32 + mt * 16 + ksel * 8;  // row within tile (VGPR r adds 0..7)
        #pragma unroll
        for (int r = 0; r < 8; ++r) {
            const float  shv    = shL[mloc + r];
            const size_t rowoff = (size_t)(m0 + mloc + r) * TDIM;
            #pragma unroll
            for (int nt = 0; nt < 4; ++nt) {
                const int n = n0 + wn * 64 + nt * 16 + lr;
                outb[rowoff + n] = acc[mt][nt][r] + shv + su_v[nt];
            }
        }
    }
}

extern "C" void kernel_launch(void* const* d_in, const int* in_sizes, int n_in,
                              void* d_out, int out_size, void* d_ws, size_t ws_size,
                              hipStream_t stream) {
    const float* H = (const float*)d_in[0];
    const float* U = (const float*)d_in[1];
    const float* W = (const float*)d_in[2];
    float* out = (float*)d_out;

    const int batches = in_sizes[0] / (TDIM * DDIM);   // 8
    dim3 grid(TDIM / TILE, TDIM / TILE, batches);
    dim3 block(256);
    size_t shmem = (size_t)2 * TILE * sizeof(float)                 // shL + suL
                 + (size_t)4 * TILE * HSTR * sizeof(_Float16);      // Hhi/Hlo/Uhi/Ulo (~136KB)
    hipLaunchKernelGGL(flow_gemm_f16split_kernel, grid, block, shmem, stream, H, U, W, out);
}
